// LinearD3Layer_42245298323619
// MI455X (gfx1250) — compile-verified, matrix operand order checked by reading, not run on hardware
//
#include <hip/hip_runtime.h>
#include <hip/hip_bf16.h>

// ---------------------------------------------------------------------------
// Problem constants (from the reference)
// ---------------------------------------------------------------------------
#define HNUM    16
#define DMODEL  1024
#define DHEAD   64
#define SLOTW   32          // SLOT
#define KDIM    16
#define NKEYS   256
#define TOPK    8
#define TOTALS  48          // TOTAL = 3*H
#define SLEN    2048
#define BSZ     4
#define NROWS   (SLEN*BSZ)  // 8192
#define QCOLS   (TOTALS*KDIM) // 768
#define LN_EPS  1e-5f

typedef __bf16 bf16;
typedef bf16  v16bf __attribute__((ext_vector_type(16)));
typedef bf16  v8bf  __attribute__((ext_vector_type(8)));
typedef float v8f   __attribute__((ext_vector_type(8)));
typedef float v2f   __attribute__((ext_vector_type(2)));

__device__ inline v8f wmma_bf16(v16bf a, v16bf b, v8f c) {
  return __builtin_amdgcn_wmma_f32_16x16x32_bf16(false, a, false, b, (short)0, c, false, false);
}
__device__ inline v8f wmma_f32(v2f a, v2f b, v8f c) {
  return __builtin_amdgcn_wmma_f32_16x16x4_f32(false, a, false, b, (short)0, c, false, false);
}
__device__ inline float elu1(float x) { return x > 0.f ? x + 1.f : __expf(x); }

// load one 16-bit A/B WMMA fragment (K-window of 32) from an LDS row pointer:
// lane L<16 needs K {0..7, 16..23}, lane L>=16 needs K {8..15, 24..31};
// caller passes row pointer already offset by half*8.
__device__ inline v16bf frag16(const bf16* rowp) {
  v8bf lo = *(const v8bf*)(rowp);
  v8bf hi = *(const v8bf*)(rowp + 16);
  v16bf f;
#pragma unroll
  for (int i = 0; i < 8; i++) { f[i] = lo[i]; f[i + 8] = hi[i]; }
  return f;
}

// ---------------------------------------------------------------------------
// K0: precision conversion + key normalization
// ---------------------------------------------------------------------------
__global__ void prep_kernel(const float* __restrict__ h, const float* __restrict__ Wq,
                            const float* __restrict__ Wo, const float* __restrict__ keys,
                            bf16* __restrict__ hbf, bf16* __restrict__ wqbf,
                            bf16* __restrict__ wobf, float* __restrict__ keyn) {
  int tid = blockIdx.x * blockDim.x + threadIdx.x;
  int nth = gridDim.x * blockDim.x;
  for (int i = tid; i < NROWS * DMODEL; i += nth) hbf[i] = (bf16)h[i];
  for (int i = tid; i < QCOLS * DMODEL; i += nth) wqbf[i] = (bf16)Wq[i];
  for (int i = tid; i < DMODEL * DMODEL; i += nth) wobf[i] = (bf16)Wo[i];
  for (int i = tid; i < 32 * NKEYS; i += nth) {
    const float* kp = keys + (size_t)i * KDIM;
    float s = 0.f;
#pragma unroll
    for (int d = 0; d < KDIM; d++) s += kp[d] * kp[d];
    float inv = rsqrtf(s);
    float* op = keyn + (size_t)i * KDIM;
#pragma unroll
    for (int d = 0; d < KDIM; d++) op[d] = kp[d] * inv;
  }
}

// ---------------------------------------------------------------------------
// K1/K4: C[M,N] = A[M,K](bf16,row) @ B[N,K](bf16,row)^T.
//   Block = 128 threads (4 waves) -> 64x64 output tile.
//   K-slab = 64, staged in LDS (double-buffered) via GLOBAL_LOAD_ASYNC_TO_LDS_B128
//   (ASYNCcnt-tracked); pipeline: issue slab k+1, s_wait_asynccnt 8 (only the
//   older slab), barrier, load ALL fragments, then 8 back-to-back
//   v_wmma_f32_16x16x32_bf16, barrier.
//   Wave w owns M-stripe [w*16, w*16+16) x all 64 N (4 accumulators).
//   mode 0: +bias then LayerNorm over each 16-col group (tile == LN group).
// ---------------------------------------------------------------------------
__global__ __launch_bounds__(128) void gemm_bf16_kernel(
    const bf16* __restrict__ A, const bf16* __restrict__ B, float* __restrict__ C,
    int M, int N, int K,
    const float* __restrict__ bias, const float* __restrict__ lng,
    const float* __restrict__ lnb, int mode) {
  __shared__ bf16 Ab[2][64][64];
  __shared__ bf16 Bb[2][64][64];

  int tid = threadIdx.x;
  int lane = tid & 31, wave = tid >> 5;
  int half = lane >> 4, rl = lane & 15;
  int m0 = blockIdx.x * 64;
  int n0 = blockIdx.y * 64;

  const bf16* Ablk = A + (size_t)m0 * K;
  const bf16* Bblk = B + (size_t)n0 * K;

  // copy plan: 2 threads per row, each thread moves 64B (4 x b128) of A and B
  int crow = tid >> 1;          // 0..63
  int cof = (tid & 1) * 32;     // element offset within the 64-wide slab

  auto issue = [&](int kc) {
    int buf = kc & 1;
    const bf16* ga = Ablk + (size_t)crow * K + kc * 64 + cof;
    const bf16* gb = Bblk + (size_t)crow * K + kc * 64 + cof;
    unsigned la = (unsigned)(size_t)&Ab[buf][crow][cof];
    unsigned lb = (unsigned)(size_t)&Bb[buf][crow][cof];
    asm volatile(
        "global_load_async_to_lds_b128 %0, %2, off\n\t"
        "global_load_async_to_lds_b128 %0, %2, off offset:16\n\t"
        "global_load_async_to_lds_b128 %0, %2, off offset:32\n\t"
        "global_load_async_to_lds_b128 %0, %2, off offset:48\n\t"
        "global_load_async_to_lds_b128 %1, %3, off\n\t"
        "global_load_async_to_lds_b128 %1, %3, off offset:16\n\t"
        "global_load_async_to_lds_b128 %1, %3, off offset:32\n\t"
        "global_load_async_to_lds_b128 %1, %3, off offset:48"
        :: "v"(la), "v"(lb), "v"(ga), "v"(gb) : "memory");
  };

  v8f acc[4];
  v8f zero = {};
#pragma unroll
  for (int nt = 0; nt < 4; nt++) acc[nt] = zero;

  int nk = K / 64;
  issue(0);
  for (int kc = 0; kc < nk; kc++) {
    int buf = kc & 1;
    if (kc + 1 < nk) {
      issue(kc + 1);
      asm volatile("s_wait_asynccnt 0x8" ::: "memory");   // older slab complete
    } else {
      asm volatile("s_wait_asynccnt 0x0" ::: "memory");
    }
    __syncthreads();

    // gather all fragments first so the 8 WMMAs can issue back-to-back
    v16bf av[2], bv[2][4];
#pragma unroll
    for (int s = 0; s < 2; s++) {
      av[s] = frag16(&Ab[buf][wave * 16 + rl][s * 32 + half * 8]);
#pragma unroll
      for (int nt = 0; nt < 4; nt++)
        bv[s][nt] = frag16(&Bb[buf][nt * 16 + rl][s * 32 + half * 8]);
    }
#pragma unroll
    for (int s = 0; s < 2; s++)
#pragma unroll
      for (int nt = 0; nt < 4; nt++)
        acc[nt] = wmma_bf16(av[s], bv[s][nt], acc[nt]);

    __syncthreads();
  }

  // C layout: element r of acc[nt] sits at (M = r + 8*half, N = rl) of the tile.
#pragma unroll
  for (int nt = 0; nt < 4; nt++) {
    int coln = n0 + nt * 16 + rl;
    if (mode == 0) {
      float gg = lng[rl], bb2 = lnb[rl];
      float bcol = bias ? bias[coln] : 0.f;
#pragma unroll
      for (int r = 0; r < 8; r++) {
        float x = acc[nt][r] + bcol;
        float s = x;
#pragma unroll
        for (int o = 1; o < 16; o <<= 1) s += __shfl_xor(s, o, 16);
        float m = s * (1.f / 16.f);
        float dd = x - m;
        float v = dd * dd;
#pragma unroll
        for (int o = 1; o < 16; o <<= 1) v += __shfl_xor(v, o, 16);
        v *= (1.f / 16.f);
        float y = dd * rsqrtf(v + LN_EPS) * gg + bb2;
        int row = m0 + wave * 16 + r + half * 8;
        C[(size_t)row * N + coln] = y;
      }
    } else {
#pragma unroll
      for (int r = 0; r < 8; r++) {
        int row = m0 + wave * 16 + r + half * 8;
        C[(size_t)row * N + coln] = acc[nt][r];
      }
    }
  }
}

// ---------------------------------------------------------------------------
// K2: PKM memory lookup, fully fused. grid=(512,48), block=32 (one wave).
//   scores(16x256, f32 WMMA K=16) -> LDS -> per-row top-8 insertion sort +
//   softmax -> sparse value gather + q@Wres^T -> @Wmem^T (f32 WMMA) -> mem.
// ---------------------------------------------------------------------------
__global__ __launch_bounds__(32) void pkm_kernel(
    const float* __restrict__ q, const float* __restrict__ keyn,
    const float* __restrict__ values, const float* __restrict__ Wres,
    const float* __restrict__ bres, const float* __restrict__ Wmem,
    const float* __restrict__ bmem, float* __restrict__ mem) {
  __shared__ float sc[16][257];
  __shared__ float pj[16][TOPK];
  __shared__ int   ij[16][TOPK];
  __shared__ float so[16][SLOTW];

  int lane = threadIdx.x;
  int half = lane >> 4, rl = lane & 15;
  int rb = blockIdx.x;          // 16-row block
  int n  = blockIdx.y;          // slot 0..47
  int np = (n < 32) ? n : (n - 32);

  // ---- scores = q . key^T (K=16) ----
  const float* qrow = q + ((size_t)(rb * 16 + rl) * TOTALS + n) * KDIM;
  for (int nt = 0; nt < 16; nt++) {
    v8f c = {};
    const float* kcol = keyn + ((size_t)np * NKEYS + nt * 16 + rl) * KDIM;
#pragma unroll
    for (int k0 = 0; k0 < 16; k0 += 4) {
      v2f a = *(const v2f*)(qrow + k0 + 2 * half);
      v2f b = *(const v2f*)(kcol + k0 + 2 * half);
      c = wmma_f32(a, b, c);
    }
#pragma unroll
    for (int r = 0; r < 8; r++) sc[r + half * 8][nt * 16 + rl] = c[r];
  }
  __syncthreads();

  // ---- top-8 + softmax per row (lanes 0..15, one row each) ----
  if (lane < 16) {
    float tv[TOPK]; int tx[TOPK];
#pragma unroll
    for (int j = 0; j < TOPK; j++) { tv[j] = -1e30f; tx[j] = 0; }
    for (int k = 0; k < NKEYS; k++) {
      float v = sc[lane][k];
      if (v > tv[TOPK - 1]) {
        int j = TOPK - 1;
        while (j > 0 && v > tv[j - 1]) { tv[j] = tv[j - 1]; tx[j] = tx[j - 1]; j--; }
        tv[j] = v; tx[j] = k;
      }
    }
    float m = tv[0], s = 0.f, e[TOPK];
#pragma unroll
    for (int j = 0; j < TOPK; j++) { e[j] = __expf(tv[j] - m); s += e[j]; }
    float inv = 1.f / s;
#pragma unroll
    for (int j = 0; j < TOPK; j++) { pj[lane][j] = e[j] * inv; ij[lane][j] = tx[j]; }
  }
  __syncthreads();

  // ---- sparse value gather + q@Wres^T + bres -> so[16][32] ----
  {
    int row = lane >> 1, c0 = (lane & 1) * 16;
    float acc[16];
#pragma unroll
    for (int c = 0; c < 16; c++) acc[c] = bres[c0 + c];
    for (int j = 0; j < TOPK; j++) {
      float p = pj[row][j];
      const float* vp = values + ((size_t)np * NKEYS + ij[row][j]) * SLOTW + c0;
#pragma unroll
      for (int c = 0; c < 16; c++) acc[c] += p * vp[c];
    }
    const float* qr = q + ((size_t)(rb * 16 + row) * TOTALS + n) * KDIM;
    float ql[16];
#pragma unroll
    for (int d = 0; d < 16; d++) ql[d] = qr[d];
#pragma unroll
    for (int c = 0; c < 16; c++) {
      const float* wr = Wres + (size_t)(c0 + c) * KDIM;
      float s2 = 0.f;
#pragma unroll
      for (int d = 0; d < 16; d++) s2 += ql[d] * wr[d];
      so[row][c0 + c] = acc[c] + s2;
    }
  }
  __syncthreads();

  // ---- out2[16][64] = so @ Wmem^T + bmem (f32 WMMA, K=32) ----
  for (int nt2 = 0; nt2 < 4; nt2++) {
    v8f c = {};
#pragma unroll
    for (int k0 = 0; k0 < 32; k0 += 4) {
      v2f a; a.x = so[rl][k0 + 2 * half]; a.y = so[rl][k0 + 2 * half + 1];
      v2f b = *(const v2f*)(Wmem + (size_t)(nt2 * 16 + rl) * SLOTW + k0 + 2 * half);
      c = wmma_f32(a, b, c);
    }
    int col = nt2 * 16 + rl;
    float bb = bmem[col];
#pragma unroll
    for (int r = 0; r < 8; r++) {
      int grow = rb * 16 + r + half * 8;
      mem[(size_t)grow * (TOTALS * DHEAD) + n * DHEAD + col] = c[r] + bb;
    }
  }
}

// ---------------------------------------------------------------------------
// K3: chunked causal linear attention. 64 blocks = (b,h); 4 waves; chunk=32.
//   Per chunk: A=QK^T (masked, WMMA), O=A@V + Q@S_prev, denom = rowsum(A)+hq.sk,
//   S += K^T V, sk += colsum(K). All matrix work on v_wmma_f32_16x16x4_f32.
// ---------------------------------------------------------------------------
__global__ __launch_bounds__(128) void linattn_kernel(const float* __restrict__ mem,
                                                      bf16* __restrict__ lo) {
  __shared__ float Qc[32][66];
  __shared__ float Kc[32][66];
  __shared__ float Vc[32][66];
  __shared__ float Am[32][34];
  __shared__ float S[64][66];
  __shared__ float sk[64];
  __shared__ float den[32];

  int tid = threadIdx.x;
  int lane = tid & 31, wave = tid >> 5;
  int half = lane >> 4, rl = lane & 15;
  int b = blockIdx.x >> 4, hh = blockIdx.x & 15;

  for (int i = tid; i < 64 * 66; i += 128) (&S[0][0])[i] = 0.f;
  for (int i = tid; i < 64; i += 128) sk[i] = 0.f;
  __syncthreads();

  const size_t baseq = (size_t)(2 * HNUM + hh) * DHEAD;
  const size_t basek = (size_t)hh * DHEAD;
  const size_t basev = (size_t)(HNUM + hh) * DHEAD;

  for (int ch = 0; ch < SLEN / 32; ch++) {
    int s0 = ch * 32;
    // load chunk, apply elu+1 to q,k
    for (int i = tid; i < 32 * 64; i += 128) {
      int t = i >> 6, d = i & 63;
      size_t row = (size_t)((s0 + t) * BSZ + b) * (TOTALS * DHEAD);
      Qc[t][d] = elu1(mem[row + baseq + d]);
      Kc[t][d] = elu1(mem[row + basek + d]);
      Vc[t][d] = mem[row + basev + d];
    }
    __syncthreads();

    // A = Q K^T (K=64), causal-masked into Am
    {
      int mt = wave >> 1, nt = wave & 1;
      v8f c = {};
#pragma unroll
      for (int k0 = 0; k0 < 64; k0 += 4) {
        v2f a; a.x = Qc[mt * 16 + rl][k0 + 2 * half]; a.y = Qc[mt * 16 + rl][k0 + 2 * half + 1];
        v2f bb; bb.x = Kc[nt * 16 + rl][k0 + 2 * half]; bb.y = Kc[nt * 16 + rl][k0 + 2 * half + 1];
        c = wmma_f32(a, bb, c);
      }
#pragma unroll
      for (int r = 0; r < 8; r++) {
        int i2 = mt * 16 + r + half * 8;
        int j2 = nt * 16 + rl;
        Am[i2][j2] = (j2 <= i2) ? c[r] : 0.f;
      }
    }
    __syncthreads();

    // O = Am@V (K=32) + Q@S_prev (K=64); 8 tiles over 4 waves (2 each)
    v8f oc[2];
    int mt2 = wave >> 1;
#pragma unroll
    for (int tt = 0; tt < 2; tt++) {
      int nt3 = (wave & 1) * 2 + tt;
      v8f c = {};
#pragma unroll
      for (int k0 = 0; k0 < 32; k0 += 4) {
        v2f a; a.x = Am[mt2 * 16 + rl][k0 + 2 * half]; a.y = Am[mt2 * 16 + rl][k0 + 2 * half + 1];
        v2f bb; bb.x = Vc[k0 + 2 * half][nt3 * 16 + rl]; bb.y = Vc[k0 + 2 * half + 1][nt3 * 16 + rl];
        c = wmma_f32(a, bb, c);
      }
#pragma unroll
      for (int k0 = 0; k0 < 64; k0 += 4) {
        v2f a; a.x = Qc[mt2 * 16 + rl][k0 + 2 * half]; a.y = Qc[mt2 * 16 + rl][k0 + 2 * half + 1];
        v2f bb; bb.x = S[k0 + 2 * half][nt3 * 16 + rl]; bb.y = S[k0 + 2 * half + 1][nt3 * 16 + rl];
        c = wmma_f32(a, bb, c);
      }
      oc[tt] = c;
    }
    // denominator: rowsum of masked A + hq . sk_prev
    if (tid < 32) {
      float s = 0.f;
      for (int j = 0; j < 32; j++) s += Am[tid][j];
      for (int d = 0; d < 64; d++) s += Qc[tid][d] * sk[d];
      den[tid] = s;
    }
    __syncthreads();

    // S += K^T V (wave owns dk-stripe wave*16..+16, 4 n-tiles)
#pragma unroll
    for (int nt4 = 0; nt4 < 4; nt4++) {
      v8f c = {};
#pragma unroll
      for (int k0 = 0; k0 < 32; k0 += 4) {
        v2f a; a.x = Kc[k0 + 2 * half][wave * 16 + rl]; a.y = Kc[k0 + 2 * half + 1][wave * 16 + rl];
        v2f bb; bb.x = Vc[k0 + 2 * half][nt4 * 16 + rl]; bb.y = Vc[k0 + 2 * half + 1][nt4 * 16 + rl];
        c = wmma_f32(a, bb, c);
      }
#pragma unroll
      for (int r = 0; r < 8; r++) S[wave * 16 + r + half * 8][nt4 * 16 + rl] += c[r];
    }
    // sk += colsum(K chunk)
    if (tid < 64) {
      float s = sk[tid];
      for (int t = 0; t < 32; t++) s += Kc[t][tid];
      sk[tid] = s;
    }
    // store O scaled by (1/sqrt(64)) / (denom + eps)
#pragma unroll
    for (int tt = 0; tt < 2; tt++) {
      int nt3 = (wave & 1) * 2 + tt;
#pragma unroll
      for (int r = 0; r < 8; r++) {
        int i2 = mt2 * 16 + r + half * 8;
        int dv = nt3 * 16 + rl;
        float scale = 0.125f / (den[i2] + 1e-5f);
        int row = (s0 + i2) * BSZ + b;
        lo[(size_t)row * DMODEL + hh * DHEAD + dv] = (bf16)(oc[tt][r] * scale);
      }
    }
    __syncthreads();
  }
}

// ---------------------------------------------------------------------------
// K5: out = LN(h + attn) over 1024, one row per block
// ---------------------------------------------------------------------------
__global__ __launch_bounds__(256) void final_ln_kernel(
    const float* __restrict__ h, const float* __restrict__ attn,
    const float* __restrict__ g, const float* __restrict__ bb,
    float* __restrict__ out) {
  __shared__ float red[8];
  __shared__ float stat[2];
  int row = blockIdx.x;
  int tid = threadIdx.x;
  const float* hr = h + (size_t)row * DMODEL;
  const float* ar = attn + (size_t)row * DMODEL;
  float y[4];
  float s = 0.f;
#pragma unroll
  for (int i = 0; i < 4; i++) { int c = tid + i * 256; y[i] = hr[c] + ar[c]; s += y[i]; }
#pragma unroll
  for (int o = 16; o >= 1; o >>= 1) s += __shfl_xor(s, o, 32);
  if ((tid & 31) == 0) red[tid >> 5] = s;
  __syncthreads();
  if (tid == 0) { float t = 0; for (int i = 0; i < 8; i++) t += red[i]; stat[0] = t * (1.f / DMODEL); }
  __syncthreads();
  float m = stat[0];
  float v = 0.f;
#pragma unroll
  for (int i = 0; i < 4; i++) { float d = y[i] - m; v += d * d; }
#pragma unroll
  for (int o = 16; o >= 1; o >>= 1) v += __shfl_xor(v, o, 32);
  __syncthreads();
  if ((tid & 31) == 0) red[tid >> 5] = v;
  __syncthreads();
  if (tid == 0) { float t = 0; for (int i = 0; i < 8; i++) t += red[i]; stat[1] = t * (1.f / DMODEL); }
  __syncthreads();
  float inv = rsqrtf(stat[1] + LN_EPS);
#pragma unroll
  for (int i = 0; i < 4; i++) {
    int c = tid + i * 256;
    out[(size_t)row * DMODEL + c] = (y[i] - m) * inv * g[c] + bb[c];
  }
}

// ---------------------------------------------------------------------------
// launch
// ---------------------------------------------------------------------------
extern "C" void kernel_launch(void* const* d_in, const int* in_sizes, int n_in,
                              void* d_out, int out_size, void* d_ws, size_t ws_size,
                              hipStream_t stream) {
  (void)in_sizes; (void)n_in; (void)out_size; (void)ws_size;
  const float* h      = (const float*)d_in[0];
  const float* Wq     = (const float*)d_in[1];
  const float* bq     = (const float*)d_in[2];
  const float* gq     = (const float*)d_in[3];
  const float* bq_ln  = (const float*)d_in[4];
  const float* keys   = (const float*)d_in[5];
  const float* values = (const float*)d_in[6];
  const float* Wres   = (const float*)d_in[7];
  const float* bres   = (const float*)d_in[8];
  const float* Wmem   = (const float*)d_in[9];
  const float* bmem   = (const float*)d_in[10];
  const float* Wo     = (const float*)d_in[11];
  const float* g_out  = (const float*)d_in[12];
  const float* b_out  = (const float*)d_in[13];
  float* out = (float*)d_out;

  char* p = (char*)d_ws;
  bf16*  hbf    = (bf16*)p;  p += (size_t)NROWS * DMODEL * 2;
  bf16*  wqbf   = (bf16*)p;  p += (size_t)QCOLS * DMODEL * 2;
  bf16*  wobf   = (bf16*)p;  p += (size_t)DMODEL * DMODEL * 2;
  float* keyn   = (float*)p; p += (size_t)32 * NKEYS * KDIM * 4;
  float* qbuf   = (float*)p; p += (size_t)NROWS * QCOLS * 4;
  float* membuf = (float*)p; p += (size_t)NROWS * TOTALS * DHEAD * 4;
  bf16*  lobf   = (bf16*)p;  p += (size_t)NROWS * DMODEL * 2;
  float* attn   = (float*)p; p += (size_t)NROWS * DMODEL * 4;

  prep_kernel<<<2048, 256, 0, stream>>>(h, Wq, Wo, keys, hbf, wqbf, wobf, keyn);

  // q = LN16(h @ Wq^T + bq)  [8192 x 768]
  gemm_bf16_kernel<<<dim3(NROWS / 64, QCOLS / 64), 128, 0, stream>>>(
      hbf, wqbf, qbuf, NROWS, QCOLS, DMODEL, bq, gq, bq_ln, 0);

  // PKM lookup -> mem [8192 x 48 x 64]
  pkm_kernel<<<dim3(NROWS / 16, TOTALS), 32, 0, stream>>>(
      qbuf, keyn, values, Wres, bres, Wmem, bmem, membuf);

  // chunked linear attention -> lo (bf16) [8192 x 1024]
  linattn_kernel<<<BSZ * HNUM, 128, 0, stream>>>(membuf, lobf);

  // attn = lo @ Wo^T  [8192 x 1024]
  gemm_bf16_kernel<<<dim3(NROWS / 64, DMODEL / 64), 128, 0, stream>>>(
      lobf, wobf, attn, NROWS, DMODEL, DMODEL, nullptr, nullptr, nullptr, 1);

  // out = LN(h + attn)
  final_ln_kernel<<<NROWS, 256, 0, stream>>>(h, attn, g_out, b_out, out);
}
